// NeuMIPv1SingleRes_1408749273558
// MI455X (gfx1250) — matrix operand order repaired and checked by reading, not compiled, not run on hardware
//
#include <hip/hip_runtime.h>
#include <hip/hip_bf16.h>

typedef __attribute__((ext_vector_type(16))) _Float16 v16h;
typedef __attribute__((ext_vector_type(8)))  _Float16 v8h;
typedef __attribute__((ext_vector_type(2)))  _Float16 h2;
typedef __attribute__((ext_vector_type(8)))  float    v8f;

#define RES 512
#define RMASK 511

__device__ __forceinline__ v8f wmma_f16(v16h a, v16h b, v8f c) {
    // D = A(16x32 f16) * B(32x16 f16) + C(16x16 f32)
    return __builtin_amdgcn_wmma_f32_16x16x32_f16(
        /*neg_a=*/false, a, /*neg_b=*/false, b,
        /*c_mod=*/(short)0, c, /*reuse_a=*/false, /*reuse_b=*/false);
}

// Bilinear-wrap fetch of 4 consecutive channels (cb4 = 0 -> ch0..3, 1 -> ch4..7)
// from a RES x RES x 8 float texture.
__device__ __forceinline__ void bilin4(const float* __restrict__ tex,
                                       float u, float v, int cb4, float o[4]) {
    float px = u * (float)RES - 0.5f;
    float py = v * (float)RES - 0.5f;
    float fpx = floorf(px), fpy = floorf(py);
    float fx = px - fpx,    fy = py - fpy;
    int ix0 = ((int)fpx) & RMASK, iy0 = ((int)fpy) & RMASK;
    int ix1 = (ix0 + 1) & RMASK,  iy1 = (iy0 + 1) & RMASK;
    const float4* t = (const float4*)tex;   // texel = 2 x float4
    float4 v00 = t[(((iy0 << 9) + ix0) << 1) + cb4];
    float4 v01 = t[(((iy0 << 9) + ix1) << 1) + cb4];
    float4 v10 = t[(((iy1 << 9) + ix0) << 1) + cb4];
    float4 v11 = t[(((iy1 << 9) + ix1) << 1) + cb4];
    float w00 = (1.f - fx) * (1.f - fy);
    float w01 = fx * (1.f - fy);
    float w10 = (1.f - fx) * fy;
    float w11 = fx * fy;
    o[0] = v00.x * w00 + v01.x * w01 + v10.x * w10 + v11.x * w11;
    o[1] = v00.y * w00 + v01.y * w01 + v10.y * w10 + v11.y * w11;
    o[2] = v00.z * w00 + v01.z * w01 + v10.z * w10 + v11.z * w11;
    o[3] = v00.w * w00 + v01.w * w01 + v10.w * w10 + v11.w * w11;
}

// Hidden-unit storage permutation: K position p holds neuron perm(p).
// perm(2n) = n (tile0 col n), perm(2n+1) = n+16 (tile1 col n).
// This makes the packed (tile0, tile1) f16 pair a single contiguous b32 store.
__device__ __host__ __forceinline__ int hperm(int p) {
    return (p >> 1) + ((p & 1) << 4);
}

// One hidden layer: D(16x32) = leaky(A * Wt + b), returned in A layout for next
// layer (K in interleaved hidden order; consumer weights are permuted to match).
// Wsh: padded 32x32 f16 weights (row n contiguous over k). act: per-wave 16x32 f16 slab.
__device__ __forceinline__ v16h layer32(const _Float16* __restrict__ Wsh,
                                        const float* __restrict__ bsh,
                                        v16h a, _Float16* __restrict__ act, int lane) {
    const int laneM = lane & 15;
    const int hi    = lane >> 4;
    // B operand: lane holds column n = laneM (+16 for tile 1), K = {hi*8..hi*8+7, 16+hi*8..16+hi*8+7}
    v16h B0 = __builtin_shufflevector(*(const v8h*)(Wsh + laneM * 32 + hi * 8), *(const v8h*)(Wsh + laneM * 32 + 16 + hi * 8), 0, 1, 2, 3, 4, 5, 6, 7, 8, 9, 10, 11, 12, 13, 14, 15);
    v16h B1 = __builtin_shufflevector(*(const v8h*)(Wsh + (laneM + 16) * 32 + hi * 8), *(const v8h*)(Wsh + (laneM + 16) * 32 + 16 + hi * 8), 0, 1, 2, 3, 4, 5, 6, 7, 8, 9, 10, 11, 12, 13, 14, 15);
    float b0 = bsh[laneM], b1 = bsh[laneM + 16];
    v8f c0, c1;
#pragma unroll
    for (int r = 0; r < 8; ++r) { c0[r] = b0; c1[r] = b1; }
    c0 = wmma_f16(a, B0, c0);
    c1 = wmma_f16(a, B1, c1);
    // D layout: lane holds rows (hi*8 + r), col laneM (tile0) / laneM+16 (tile1).
    // Pack pair -> f16 (v_cvt_pk_f16_f32), leaky = max(x, 0.01x) in packed f16,
    // store as ONE ds_store_b32 at interleaved K position {2*laneM, 2*laneM+1}.
    const int rowB = hi * 8;
#pragma unroll
    for (int r = 0; r < 8; ++r) {
        h2 p;
        p[0] = (_Float16)c0[r];
        p[1] = (_Float16)c1[r];
        h2 q = p * (_Float16)0.01f;
        h2 m = __builtin_elementwise_max(p, q);   // leaky ReLU, slope 0.01 < 1
        *(h2*)(act + (rowB + r) * 32 + 2 * laneM) = m;
    }
    // Reload in A layout: lanes 0-15: K {0..7,16..23} of row laneM; lanes 16-31: K {8..15,24..31}.
    // Per-wave LDS ops are in-order (DScnt), no barrier needed.
    v8h lo = *(const v8h*)(act + laneM * 32 + hi * 8);
    v8h hh = *(const v8h*)(act + laneM * 32 + 16 + hi * 8);
    v16h an;
#pragma unroll
    for (int e = 0; e < 8; ++e) { an[e] = lo[e]; an[8 + e] = hh[e]; }
    return an;
}

// Final layer (out_dim <= 16): single N-tile, returns raw D column data.
__device__ __forceinline__ v8f layer_out(const _Float16* __restrict__ Wsh,
                                         const float* __restrict__ bsh,
                                         v16h a, int lane) {
    const int laneM = lane & 15;
    const int hi    = lane >> 4;
    v16h B0 = __builtin_shufflevector(*(const v8h*)(Wsh + laneM * 32 + hi * 8), *(const v8h*)(Wsh + laneM * 32 + 16 + hi * 8), 0, 1, 2, 3, 4, 5, 6, 7, 8, 9, 10, 11, 12, 13, 14, 15);
    float b0 = bsh[laneM];
    v8f c;
#pragma unroll
    for (int r = 0; r < 8; ++r) c[r] = b0;
    return wmma_f16(a, B0, c);
}

__global__ void __launch_bounds__(256) neumip_kernel(
    const float* __restrict__ cam,  const float* __restrict__ light,
    const float* __restrict__ uv,   const float* __restrict__ offt,
    const float* __restrict__ rgbt,
    const float* __restrict__ ow0, const float* __restrict__ ob0,
    const float* __restrict__ ow1, const float* __restrict__ ob1,
    const float* __restrict__ ow2, const float* __restrict__ ob2,
    const float* __restrict__ ow3, const float* __restrict__ ob3,
    const float* __restrict__ rw0, const float* __restrict__ rb0,
    const float* __restrict__ rw1, const float* __restrict__ rb1,
    const float* __restrict__ rw2, const float* __restrict__ rb2,
    const float* __restrict__ rw3, const float* __restrict__ rb3,
    float* __restrict__ out, int nGroups)
{
    // Padded f16 weights (B-operand friendly), f32 biases, per-wave activation slabs.
    __shared__ __align__(16) _Float16 sW[8][1024];   // 16 KB
    __shared__ float                  sBias[8][32];  // 1 KB
    __shared__ __align__(16) _Float16 sAct[8][512];  // 8 KB (per-wave 16x32)
    __shared__ float                  sDepth[8][16]; // 512 B

    {
        const float* Wp[8]  = {ow0, ow1, ow2, ow3, rw0, rw1, rw2, rw3};
        const float* Bpv[8] = {ob0, ob1, ob2, ob3, rb0, rb1, rb2, rb3};
        const int nin[8]  = {10, 32, 32, 32, 12, 32, 32, 32};
        const int nout[8] = {32, 32, 32,  1, 32, 32, 32,  3};
        for (int idx = threadIdx.x; idx < 8 * 1024; idx += blockDim.x) {
            int m = idx >> 10, rc = idx & 1023, n = rc >> 5, p = rc & 31;
            // Layers consuming hidden activations (m != 0,4) use permuted k columns.
            int k = (m == 0 || m == 4) ? p : hperm(p);
            float val = (n < nout[m] && k < nin[m]) ? Wp[m][n * nin[m] + k] : 0.f;
            sW[m][rc] = (_Float16)val;
        }
        for (int idx = threadIdx.x; idx < 8 * 32; idx += blockDim.x) {
            int m = idx >> 5, n = idx & 31;
            sBias[m][n] = (n < nout[m]) ? Bpv[m][n] : 0.f;
        }
    }
    __syncthreads();

    const int lane  = threadIdx.x & 31;
    const int wv    = threadIdx.x >> 5;
    const int laneM = lane & 15;
    const int hi    = lane >> 4;
    _Float16* act = sAct[wv];
    float*    dep = sDepth[wv];

    const int wavesTotal = (gridDim.x * blockDim.x) >> 5;
    const int wave0 = (blockIdx.x * blockDim.x + threadIdx.x) >> 5;

    for (int g = wave0; g < nGroups; g += wavesTotal) {   // wave-uniform loop
        const int s = g * 16 + laneM;                     // this lane's sample row
        float cx = cam[2 * s],   cy = cam[2 * s + 1];
        float u  = uv[2 * s],    v  = uv[2 * s + 1];
        float lx = light[2 * s], ly = light[2 * s + 1];

        // ---- depth MLP input: [off_lat(8) | cam(2) | pad] as 16x32 f16 A tile ----
        v16h a;
#pragma unroll
        for (int e = 0; e < 16; ++e) a[e] = (_Float16)0.f;
        if (hi == 0) {                     // low half-wave: K0..7 = latent channels
            float l0[4], l1[4];
            bilin4(offt, u, v, 0, l0);
            bilin4(offt, u, v, 1, l1);
#pragma unroll
            for (int e = 0; e < 4; ++e) { a[e] = (_Float16)l0[e]; a[4 + e] = (_Float16)l1[e]; }
        } else {                           // high half-wave: K8,K9 = camera dir
            a[0] = (_Float16)cx; a[1] = (_Float16)cy;
        }

        a = layer32(sW[0], sBias[0], a, act, lane);
        a = layer32(sW[1], sBias[1], a, act, lane);
        a = layer32(sW[2], sBias[2], a, act, lane);
        v8f dc = layer_out(sW[3], sBias[3], a, lane);

        // depth = column 0 of final tile (lives in lanes 0 and 16); broadcast via LDS
        if (laneM == 0) {
#pragma unroll
            for (int r = 0; r < 8; ++r) dep[hi * 8 + r] = dc[r];
        }
        float depth = dep[laneM];
        float zz = fmaxf(1.f - cx * cx - cy * cy, 1e-6f);
        float z  = sqrtf(zz);
        float u2 = u + cx / z * depth;
        float v2 = v + cy / z * depth;

        // ---- rgb MLP input: [light(2) | cam(2) | rgb_lat(8) | pad] ----
        float l2[4];
        bilin4(rgbt, u2, v2, hi, l2);      // low half: ch0..3, high half: ch4..7
        v16h a2;
#pragma unroll
        for (int e = 0; e < 16; ++e) a2[e] = (_Float16)0.f;
        if (hi == 0) {
            a2[0] = (_Float16)lx;    a2[1] = (_Float16)ly;
            a2[2] = (_Float16)cx;    a2[3] = (_Float16)cy;
            a2[4] = (_Float16)l2[0]; a2[5] = (_Float16)l2[1];
            a2[6] = (_Float16)l2[2]; a2[7] = (_Float16)l2[3];
        } else {                            // K8..11 = rgb_lat[4..7]
            a2[0] = (_Float16)l2[0]; a2[1] = (_Float16)l2[1];
            a2[2] = (_Float16)l2[2]; a2[3] = (_Float16)l2[3];
        }

        a2 = layer32(sW[4], sBias[4], a2, act, lane);
        a2 = layer32(sW[5], sBias[5], a2, act, lane);
        a2 = layer32(sW[6], sBias[6], a2, act, lane);
        v8f rc = layer_out(sW[7], sBias[7], a2, lane);

        // D cols 0..2 = r,g,b; col n is held by lanes n and n+16 (rows hi*8 + r)
        if (laneM < 3) {
#pragma unroll
            for (int r = 0; r < 8; ++r)
                out[(g * 16 + hi * 8 + r) * 3 + laneM] = rc[r];
        }
    }
}

extern "C" void kernel_launch(void* const* d_in, const int* in_sizes, int n_in,
                              void* d_out, int out_size, void* d_ws, size_t ws_size,
                              hipStream_t stream) {
    const float* cam   = (const float*)d_in[0];
    const float* light = (const float*)d_in[1];
    const float* uvp   = (const float*)d_in[2];
    const float* offt  = (const float*)d_in[3];
    const float* rgbt  = (const float*)d_in[4];
    const float* ow0 = (const float*)d_in[5];  const float* ob0 = (const float*)d_in[6];
    const float* ow1 = (const float*)d_in[7];  const float* ob1 = (const float*)d_in[8];
    const float* ow2 = (const float*)d_in[9];  const float* ob2 = (const float*)d_in[10];
    const float* ow3 = (const float*)d_in[11]; const float* ob3 = (const float*)d_in[12];
    const float* rw0 = (const float*)d_in[13]; const float* rb0 = (const float*)d_in[14];
    const float* rw1 = (const float*)d_in[15]; const float* rb1 = (const float*)d_in[16];
    const float* rw2 = (const float*)d_in[17]; const float* rb2 = (const float*)d_in[18];
    const float* rw3 = (const float*)d_in[19]; const float* rb3 = (const float*)d_in[20];

    const int Bsz = in_sizes[0] / 2;      // camera_dir is (B,2)
    const int nGroups = Bsz / 16;

    dim3 grid(1024), block(256);          // 8192 persistent waves, 8 groups each
    neumip_kernel<<<grid, block, 0, stream>>>(
        cam, light, uvp, offt, rgbt,
        ow0, ob0, ow1, ob1, ow2, ob2, ow3, ob3,
        rw0, rb0, rw1, rb1, rw2, rb2, rw3, rb3,
        (float*)d_out, nGroups);
}
